// MultiHeadLatentAttention_8323646619997
// MI455X (gfx1250) — matrix operand order repaired and checked by reading, not compiled, not run on hardware
//
#include <hip/hip_runtime.h>
#include <hip/hip_bf16.h>
#include <math.h>

typedef unsigned short u16;
typedef unsigned int   u32;
typedef __attribute__((ext_vector_type(16))) __bf16 v16bf;
typedef __attribute__((ext_vector_type(8)))  __bf16 v8bf;
typedef __attribute__((ext_vector_type(8)))  float  v8f;

#define N_EMBD   1024
#define N_HEAD   16
#define HEAD_DIM 64
#define KV_RANK  256
#define Q_RANK   512
#define ROPE_DIM 32
#define QK_DIM   96
#define BATCH    2
#define SEQ      2048
#define NTOK     (BATCH*SEQ)

__device__ __forceinline__ u16 f2bf(float f) {
  u32 u = __float_as_uint(f);
  u32 r = u + 0x7FFFu + ((u >> 16) & 1u);   // round-to-nearest-even
  return (u16)(r >> 16);
}
__device__ __forceinline__ float bf2f(u16 h) {
  return __uint_as_float(((u32)h) << 16);
}

// async global -> LDS copy of 16 bytes (per-lane addresses), ASYNCcnt tracked
__device__ __forceinline__ void async_copy_b128(const u16* gsrc, u16* lds_dst) {
  u32 l = (u32)(size_t)lds_dst;          // wave-relative LDS byte offset
  asm volatile("global_load_async_to_lds_b128 %0, %1, off"
               :: "v"(l), "v"(gsrc) : "memory");
}
__device__ __forceinline__ void async_wait0() {
  asm volatile("s_wait_asynccnt 0x0" ::: "memory");
}

// ---------------------------------------------------------------- conversion
__global__ __launch_bounds__(256) void cvt_f32_bf16(const float* __restrict__ in,
                                                    u16* __restrict__ out, int n) {
  for (int i = blockIdx.x*blockDim.x + threadIdx.x; i < n; i += gridDim.x*blockDim.x)
    out[i] = f2bf(in[i]);
}

// ---------------------------------------------------------------- bf16 GEMM
// C[M,N] = A[M,K] * B[K,N]; block tile 128x128, BK=64, 8 waves (wave32),
// double-buffered LDS; A tiles via async-to-LDS, B tiles register-staged
// (transpose scatter). 16 WMMA per barrier per wave.
#define BM 128
#define BN 128
#define BK 64
#define LDA 72
#define LDBT 72

__global__ __launch_bounds__(256) void gemm_bf16(
    const u16* __restrict__ A, const u16* __restrict__ B,
    u16* __restrict__ Cb, float* __restrict__ Cf,
    int M, int N, int K)
{
  __shared__ __align__(16) u16 As[2][BM*LDA];   // [m][k]
  __shared__ __align__(16) u16 Bt[2][BN*LDBT];  // [n][k] (transposed for B-frag)

  const int tid  = threadIdx.x;
  const int lane = tid & 31, wid = tid >> 5;
  const int hf   = lane >> 4, l16 = lane & 15;
  const int wm   = (wid >> 1) * 32;
  const int wn   = (wid & 1) * 64;
  const int bm   = blockIdx.x * BM;
  const int bn   = blockIdx.y * BN;

  // per-thread chunk coords (4 x 16B chunks for A, 4 for B)
  int arr[4], akk[4], bkr[4], bnn[4];
#pragma unroll
  for (int i = 0; i < 4; ++i) {
    int c = tid*4 + i;
    arr[i] = c >> 3;        akk[i] = (c & 7) * 8;     // A: 8 chunks per 64-wide row
    bkr[i] = c >> 4;        bnn[i] = (c & 15) * 8;    // B: 16 chunks per 128-wide row
  }

  v8f zero = {};
  v8f acc[2][4];
#pragma unroll
  for (int mi = 0; mi < 2; ++mi)
#pragma unroll
    for (int ni = 0; ni < 4; ++ni) acc[mi][ni] = zero;

  v8bf breg[4];

  // prologue: tile 0
#pragma unroll
  for (int i = 0; i < 4; ++i)
    async_copy_b128(A + (size_t)(bm + arr[i])*K + akk[i], &As[0][arr[i]*LDA + akk[i]]);
#pragma unroll
  for (int i = 0; i < 4; ++i)
    breg[i] = *(const v8bf*)(B + (size_t)bkr[i]*N + bn + bnn[i]);
#pragma unroll
  for (int i = 0; i < 4; ++i) {
    u16 t[8]; *(v8bf*)t = breg[i];
#pragma unroll
    for (int j = 0; j < 8; ++j) Bt[0][(bnn[i] + j)*LDBT + bkr[i]] = t[j];
  }

  const int nk = K / BK;
  for (int kt = 0; kt < nk; ++kt) {
    async_wait0();
    __syncthreads();
    const int cur = kt & 1;
    const int k1 = (kt + 1) * BK;

    if (kt + 1 < nk) {
#pragma unroll
      for (int i = 0; i < 4; ++i)
        async_copy_b128(A + (size_t)(bm + arr[i])*K + k1 + akk[i],
                        &As[cur ^ 1][arr[i]*LDA + akk[i]]);
#pragma unroll
      for (int i = 0; i < 4; ++i)
        breg[i] = *(const v8bf*)(B + (size_t)(k1 + bkr[i])*N + bn + bnn[i]);
    }

    // compute two K=32 halves from the current buffer
#pragma unroll
    for (int ks = 0; ks < 2; ++ks) {
      v16bf af[2], bfrag[4];
#pragma unroll
      for (int mi = 0; mi < 2; ++mi) {
        const u16* p = &As[cur][(wm + mi*16 + l16)*LDA + ks*32 + hf*8];
        ((v8bf*)&af[mi])[0] = *(const v8bf*)p;
        ((v8bf*)&af[mi])[1] = *(const v8bf*)(p + 16);
      }
#pragma unroll
      for (int ni = 0; ni < 4; ++ni) {
        const u16* p = &Bt[cur][(wn + ni*16 + l16)*LDBT + ks*32 + hf*8];
        ((v8bf*)&bfrag[ni])[0] = *(const v8bf*)p;
        ((v8bf*)&bfrag[ni])[1] = *(const v8bf*)(p + 16);
      }
#pragma unroll
      for (int mi = 0; mi < 2; ++mi)
#pragma unroll
        for (int ni = 0; ni < 4; ++ni)
          acc[mi][ni] = __builtin_amdgcn_wmma_f32_16x16x32_bf16(
              false, af[mi], false, bfrag[ni], (short)0, acc[mi][ni], false, false);
    }

    if (kt + 1 < nk) {
      const int nxt = cur ^ 1;
#pragma unroll
      for (int i = 0; i < 4; ++i) {
        u16 t[8]; *(v8bf*)t = breg[i];
#pragma unroll
        for (int j = 0; j < 8; ++j) Bt[nxt][(bnn[i] + j)*LDBT + bkr[i]] = t[j];
      }
    }
  }

#pragma unroll
  for (int mi = 0; mi < 2; ++mi)
#pragma unroll
    for (int ni = 0; ni < 4; ++ni)
#pragma unroll
      for (int r = 0; r < 8; ++r) {
        int m = bm + wm + mi*16 + r + 8*hf;
        int n = bn + wn + ni*16 + l16;
        float v = acc[mi][ni][r];
        if (Cb) Cb[(size_t)m*N + n] = f2bf(v);
        if (Cf) Cf[(size_t)m*N + n] = v;
      }
}

// ---------------------------------------------------------------- RMSNorm
__global__ __launch_bounds__(256) void rmsnorm_kv(const float* __restrict__ in,
                                                  const float* __restrict__ g,
                                                  u16* __restrict__ out) {
  __shared__ float red[256];
  const int row = blockIdx.x, tid = threadIdx.x;
  float v = in[(size_t)row*KV_RANK + tid];
  red[tid] = v*v;
  __syncthreads();
  for (int s = 128; s > 0; s >>= 1) { if (tid < s) red[tid] += red[tid+s]; __syncthreads(); }
  float rs = rsqrtf(red[0]/(float)KV_RANK + 1e-6f);
  out[(size_t)row*KV_RANK + tid] = f2bf(v * rs * g[tid]);
}

// ---------------------------------------------------------------- RoPE + pack
__global__ __launch_bounds__(256) void pack_qkv(
    const u16* __restrict__ qcont, const u16* __restrict__ qrope,
    const u16* __restrict__ kcont, const u16* __restrict__ krope,
    const u16* __restrict__ vcont,
    u16* __restrict__ qpack, u16* __restrict__ kpack, u16* __restrict__ vpack)
{
  const int totalQK = NTOK * N_HEAD * QK_DIM;
  for (int idx = blockIdx.x*blockDim.x + threadIdx.x; idx < totalQK; idx += gridDim.x*blockDim.x) {
    int token = idx / (N_HEAD*QK_DIM);
    int rem   = idx % (N_HEAD*QK_DIM);
    int h = rem / QK_DIM, d = rem % QK_DIM;
    int b = token / SEQ, t = token % SEQ;
    size_t dst = (((size_t)(b*N_HEAD + h)*SEQ) + t)*QK_DIM + d;
    float qv, kvv;
    if (d < HEAD_DIM) {
      qv  = bf2f(qcont[(size_t)token*N_EMBD + h*HEAD_DIM + d]);
      kvv = bf2f(kcont[(size_t)token*N_EMBD + h*HEAD_DIM + d]);
    } else {
      int i = d - HEAD_DIM;
      int j = i & 15;
      float invf = __powf(10000.0f, -(float)(2*j)/(float)ROPE_DIM);
      float ang = (float)t * invf;
      float c = __cosf(ang), sn = __sinf(ang);
      size_t base = (size_t)token*(N_HEAD*ROPE_DIM) + (size_t)h*ROPE_DIM;
      float xq = bf2f(qrope[base + i]);
      float xk = bf2f(krope[base + i]);
      float rq = (i < 16) ? -bf2f(qrope[base + i + 16]) : bf2f(qrope[base + i - 16]);
      float rk = (i < 16) ? -bf2f(krope[base + i + 16]) : bf2f(krope[base + i - 16]);
      qv  = xq*c + rq*sn;
      kvv = xk*c + rk*sn;
    }
    qpack[dst] = f2bf(qv);
    kpack[dst] = f2bf(kvv);
  }
  const int totalV = NTOK * N_HEAD * HEAD_DIM;
  for (int idx = blockIdx.x*blockDim.x + threadIdx.x; idx < totalV; idx += gridDim.x*blockDim.x) {
    int token = idx / (N_HEAD*HEAD_DIM);
    int rem   = idx % (N_HEAD*HEAD_DIM);
    int h = rem / HEAD_DIM, d = rem % HEAD_DIM;
    int b = token / SEQ, t = token % SEQ;
    vpack[(((size_t)(b*N_HEAD + h)*SEQ) + t)*HEAD_DIM + d] =
        vcont[(size_t)token*N_EMBD + h*HEAD_DIM + d];
  }
}

// ---------------------------------------------------------------- flash attention
// grid: (SEQ/64, BATCH*N_HEAD), block 128 (4 waves, 16 q-rows each)
// Q/K tiles via async-to-LDS; K/V double-buffered.
__global__ __launch_bounds__(128) void mla_flash_attn(
    const u16* __restrict__ Q, const u16* __restrict__ Kp, const u16* __restrict__ Vp,
    u16* __restrict__ Oout)
{
  __shared__ __align__(16) u16 Qs[64*104];
  __shared__ __align__(16) u16 Kt[2][64*104];
  __shared__ __align__(16) u16 Vt[2][64*72];   // [d][key]
  __shared__ __align__(16) u16 Ps[64*72];      // [qrow][key]

  const int tid  = threadIdx.x;
  const int lane = tid & 31;
  const int wid  = tid >> 5;
  const int hf   = lane >> 4;
  const int l16  = lane & 15;
  const int wrow = wid * 16;

  const int bh = blockIdx.y;
  const int qb = blockIdx.x * 64;

  // async-load Q tile
  const size_t qoff = ((size_t)bh * SEQ + qb) * QK_DIM;
#pragma unroll
  for (int i = 0; i < 6; ++i) {
    int c = i*128 + tid, r = c / 12, kc = (c % 12) * 8;
    async_copy_b128(Q + qoff + (size_t)r*QK_DIM + kc, &Qs[r*104 + kc]);
  }

  float m_i[8], l_i[8];
  v8f zero = {};
  v8f oacc[4];
#pragma unroll
  for (int r = 0; r < 8; ++r) { m_i[r] = -1e30f; l_i[r] = 0.f; }
#pragma unroll
  for (int nf = 0; nf < 4; ++nf) oacc[nf] = zero;

  const float scale = 0.10206207261596575f;  // 1/sqrt(96)

  v8bf vreg[4];

  // prologue: key-block 0 (K async, V register-staged transpose)
  {
    const size_t koff = (size_t)bh * SEQ * QK_DIM;
    const size_t voff = (size_t)bh * SEQ * HEAD_DIM;
#pragma unroll
    for (int i = 0; i < 6; ++i) {
      int c = i*128 + tid, r = c / 12, kc = (c % 12) * 8;
      async_copy_b128(Kp + koff + (size_t)r*QK_DIM + kc, &Kt[0][r*104 + kc]);
    }
#pragma unroll
    for (int i = 0; i < 4; ++i) {
      int c = i*128 + tid, r = c / 8, dc = (c % 8) * 8;
      vreg[i] = *(const v8bf*)(Vp + voff + (size_t)r*HEAD_DIM + dc);
    }
#pragma unroll
    for (int i = 0; i < 4; ++i) {
      int c = i*128 + tid, r = c / 8, dc = (c % 8) * 8;
      u16 tmp[8]; *(v8bf*)tmp = vreg[i];
#pragma unroll
      for (int j = 0; j < 8; ++j) Vt[0][(dc + j)*72 + r] = tmp[j];
    }
  }

  for (int kb = 0; kb <= qb; kb += 64) {
    async_wait0();
    __syncthreads();
    const int cur = (kb >> 6) & 1;
    const bool more = (kb + 64) <= qb;

    if (more) {
      const size_t koff = ((size_t)bh * SEQ + kb + 64) * QK_DIM;
      const size_t voff = ((size_t)bh * SEQ + kb + 64) * HEAD_DIM;
#pragma unroll
      for (int i = 0; i < 6; ++i) {
        int c = i*128 + tid, r = c / 12, kc = (c % 12) * 8;
        async_copy_b128(Kp + koff + (size_t)r*QK_DIM + kc, &Kt[cur ^ 1][r*104 + kc]);
      }
#pragma unroll
      for (int i = 0; i < 4; ++i) {
        int c = i*128 + tid, r = c / 8, dc = (c % 8) * 8;
        vreg[i] = *(const v8bf*)(Vp + voff + (size_t)r*HEAD_DIM + dc);
      }
    }

    // S = Q * K^T (16x64 per wave, K-dim 96)
    v8f s[4];
#pragma unroll
    for (int nf = 0; nf < 4; ++nf) s[nf] = zero;
#pragma unroll
    for (int ks = 0; ks < 3; ++ks) {
      v16bf a;
      const u16* ap = &Qs[(wrow + l16)*104 + ks*32 + hf*8];
      ((v8bf*)&a)[0] = *(const v8bf*)ap;
      ((v8bf*)&a)[1] = *(const v8bf*)(ap + 16);
#pragma unroll
      for (int nf = 0; nf < 4; ++nf) {
        v16bf bm;
        const u16* bp = &Kt[cur][(nf*16 + l16)*104 + ks*32 + hf*8];
        ((v8bf*)&bm)[0] = *(const v8bf*)bp;
        ((v8bf*)&bm)[1] = *(const v8bf*)(bp + 16);
        s[nf] = __builtin_amdgcn_wmma_f32_16x16x32_bf16(false, a, false, bm,
                                                        (short)0, s[nf], false, false);
      }
    }

    // scale + causal mask
#pragma unroll
    for (int nf = 0; nf < 4; ++nf)
#pragma unroll
      for (int r = 0; r < 8; ++r) {
        float v = s[nf][r] * scale;
        int qrow = qb + wrow + r + 8*hf;
        int krow = kb + nf*16 + l16;
        s[nf][r] = (krow <= qrow) ? v : -1e30f;
      }

    // online softmax (row reductions across the 16-lane half via shfl_xor)
    float fac[8];
#pragma unroll
    for (int r = 0; r < 8; ++r) {
      float mx = s[0][r];
#pragma unroll
      for (int nf = 1; nf < 4; ++nf) mx = fmaxf(mx, s[nf][r]);
#pragma unroll
      for (int off = 1; off < 16; off <<= 1) mx = fmaxf(mx, __shfl_xor(mx, off, 32));
      float mnew = fmaxf(m_i[r], mx);
      fac[r] = __expf(m_i[r] - mnew);
      float rs = 0.f;
#pragma unroll
      for (int nf = 0; nf < 4; ++nf) {
        float p = __expf(s[nf][r] - mnew);
        rs += p;
        Ps[(wrow + r + 8*hf)*72 + nf*16 + l16] = f2bf(p);
      }
#pragma unroll
      for (int off = 1; off < 16; off <<= 1) rs += __shfl_xor(rs, off, 32);
      l_i[r] = l_i[r]*fac[r] + rs;
      m_i[r] = mnew;
    }

#pragma unroll
    for (int nf = 0; nf < 4; ++nf)
#pragma unroll
      for (int r = 0; r < 8; ++r) oacc[nf][r] *= fac[r];

    // O += P * V (K-dim 64 over keys)
#pragma unroll
    for (int ks = 0; ks < 2; ++ks) {
      v16bf a;
      const u16* ap = &Ps[(wrow + l16)*72 + ks*32 + hf*8];
      ((v8bf*)&a)[0] = *(const v8bf*)ap;
      ((v8bf*)&a)[1] = *(const v8bf*)(ap + 16);
#pragma unroll
      for (int nf = 0; nf < 4; ++nf) {
        v16bf bm;
        const u16* bp = &Vt[cur][(nf*16 + l16)*72 + ks*32 + hf*8];
        ((v8bf*)&bm)[0] = *(const v8bf*)bp;
        ((v8bf*)&bm)[1] = *(const v8bf*)(bp + 16);
        oacc[nf] = __builtin_amdgcn_wmma_f32_16x16x32_bf16(false, a, false, bm,
                                                           (short)0, oacc[nf], false, false);
      }
    }

    // stage next V tile into the other buffer
    if (more) {
      const int nxt = cur ^ 1;
#pragma unroll
      for (int i = 0; i < 4; ++i) {
        int c = i*128 + tid, r = c / 8, dc = (c % 8) * 8;
        u16 tmp[8]; *(v8bf*)tmp = vreg[i];
#pragma unroll
        for (int j = 0; j < 8; ++j) Vt[nxt][(dc + j)*72 + r] = tmp[j];
      }
    }
  }

  const int b = bh / N_HEAD, h = bh % N_HEAD;
#pragma unroll
  for (int r = 0; r < 8; ++r) {
    float inv = 1.0f / l_i[r];
    int token = b*SEQ + qb + wrow + r + 8*hf;
#pragma unroll
    for (int nf = 0; nf < 4; ++nf) {
      int d = nf*16 + l16;
      Oout[(size_t)token*N_EMBD + h*HEAD_DIM + d] = f2bf(oacc[nf][r] * inv);
    }
  }
}

// ---------------------------------------------------------------- launch
extern "C" void kernel_launch(void* const* d_in, const int* in_sizes, int n_in,
                              void* d_out, int out_size, void* d_ws, size_t ws_size,
                              hipStream_t stream)
{
  const float* x       = (const float*)d_in[0];
  const float* W_kv    = (const float*)d_in[1];
  const float* g_kv    = (const float*)d_in[2];
  const float* W_kdec  = (const float*)d_in[3];
  const float* W_vdec  = (const float*)d_in[4];
  const float* W_q     = (const float*)d_in[5];
  const float* W_qdec  = (const float*)d_in[6];
  const float* W_krope = (const float*)d_in[7];
  const float* W_qrope = (const float*)d_in[8];
  const float* W_o     = (const float*)d_in[9];
  float* out = (float*)d_out;

  char* ws = (char*)d_ws;
  size_t off = 0;
  auto alloc = [&](size_t bytes) -> char* {
    char* p = ws + off;
    off += (bytes + 255) & ~(size_t)255;
    return p;
  };

  u16*   xb     = (u16*)  alloc((size_t)NTOK*N_EMBD*2);
  u16*   wkv    = (u16*)  alloc((size_t)N_EMBD*KV_RANK*2);
  u16*   wq     = (u16*)  alloc((size_t)N_EMBD*Q_RANK*2);
  u16*   wkdec  = (u16*)  alloc((size_t)KV_RANK*N_EMBD*2);
  u16*   wvdec  = (u16*)  alloc((size_t)KV_RANK*N_EMBD*2);
  u16*   wqdec  = (u16*)  alloc((size_t)Q_RANK*N_EMBD*2);
  u16*   wkrope = (u16*)  alloc((size_t)N_EMBD*(N_HEAD*ROPE_DIM)*2);
  u16*   wqrope = (u16*)  alloc((size_t)Q_RANK*(N_HEAD*ROPE_DIM)*2);
  u16*   wo     = (u16*)  alloc((size_t)N_EMBD*N_EMBD*2);
  float* kvf    = (float*)alloc((size_t)NTOK*KV_RANK*4);
  u16*   kvb    = (u16*)  alloc((size_t)NTOK*KV_RANK*2);
  u16*   qcb    = (u16*)  alloc((size_t)NTOK*Q_RANK*2);
  u16*   kcont  = (u16*)  alloc((size_t)NTOK*N_EMBD*2);
  u16*   vout   = (u16*)  alloc((size_t)NTOK*N_EMBD*2);
  u16*   qcont  = (u16*)  alloc((size_t)NTOK*N_EMBD*2);
  u16*   krr    = (u16*)  alloc((size_t)NTOK*(N_HEAD*ROPE_DIM)*2);
  u16*   qrr    = (u16*)  alloc((size_t)NTOK*(N_HEAD*ROPE_DIM)*2);
  u16*   qpack  = (u16*)  alloc((size_t)BATCH*N_HEAD*SEQ*QK_DIM*2);
  u16*   kpack  = (u16*)  alloc((size_t)BATCH*N_HEAD*SEQ*QK_DIM*2);
  u16*   vpack  = (u16*)  alloc((size_t)BATCH*N_HEAD*SEQ*HEAD_DIM*2);
  u16*   attnout= (u16*)  alloc((size_t)NTOK*N_EMBD*2);
  (void)ws_size; (void)in_sizes; (void)n_in; (void)out_size;

  // ---- bf16 conversions
  cvt_f32_bf16<<<2048, 256, 0, stream>>>(x,       xb,     NTOK*N_EMBD);
  cvt_f32_bf16<<<512,  256, 0, stream>>>(W_kv,    wkv,    N_EMBD*KV_RANK);
  cvt_f32_bf16<<<512,  256, 0, stream>>>(W_q,     wq,     N_EMBD*Q_RANK);
  cvt_f32_bf16<<<512,  256, 0, stream>>>(W_kdec,  wkdec,  KV_RANK*N_EMBD);
  cvt_f32_bf16<<<512,  256, 0, stream>>>(W_vdec,  wvdec,  KV_RANK*N_EMBD);
  cvt_f32_bf16<<<512,  256, 0, stream>>>(W_qdec,  wqdec,  Q_RANK*N_EMBD);
  cvt_f32_bf16<<<512,  256, 0, stream>>>(W_krope, wkrope, N_EMBD*N_HEAD*ROPE_DIM);
  cvt_f32_bf16<<<512,  256, 0, stream>>>(W_qrope, wqrope, Q_RANK*N_HEAD*ROPE_DIM);
  cvt_f32_bf16<<<1024, 256, 0, stream>>>(W_o,     wo,     N_EMBD*N_EMBD);

  // ---- projections (all WMMA bf16 GEMMs)
  gemm_bf16<<<dim3(NTOK/BM, KV_RANK/BN), 256, 0, stream>>>(xb, wkv, nullptr, kvf,
                                                           NTOK, KV_RANK, N_EMBD);
  rmsnorm_kv<<<NTOK, 256, 0, stream>>>(kvf, g_kv, kvb);
  gemm_bf16<<<dim3(NTOK/BM, Q_RANK/BN), 256, 0, stream>>>(xb, wq, qcb, nullptr,
                                                          NTOK, Q_RANK, N_EMBD);
  gemm_bf16<<<dim3(NTOK/BM, N_EMBD/BN), 256, 0, stream>>>(kvb, wkdec, kcont, nullptr,
                                                          NTOK, N_EMBD, KV_RANK);
  gemm_bf16<<<dim3(NTOK/BM, N_EMBD/BN), 256, 0, stream>>>(kvb, wvdec, vout, nullptr,
                                                          NTOK, N_EMBD, KV_RANK);
  gemm_bf16<<<dim3(NTOK/BM, N_EMBD/BN), 256, 0, stream>>>(qcb, wqdec, qcont, nullptr,
                                                          NTOK, N_EMBD, Q_RANK);
  gemm_bf16<<<dim3(NTOK/BM, (N_HEAD*ROPE_DIM)/BN), 256, 0, stream>>>(xb, wkrope, krr, nullptr,
                                                          NTOK, N_HEAD*ROPE_DIM, N_EMBD);
  gemm_bf16<<<dim3(NTOK/BM, (N_HEAD*ROPE_DIM)/BN), 256, 0, stream>>>(qcb, wqrope, qrr, nullptr,
                                                          NTOK, N_HEAD*ROPE_DIM, Q_RANK);

  // ---- RoPE + head-major packing
  pack_qkv<<<4096, 256, 0, stream>>>(qcont, qrr, kcont, krr, vout, qpack, kpack, vpack);

  // ---- flash attention
  mla_flash_attn<<<dim3(SEQ/64, BATCH*N_HEAD), 128, 0, stream>>>(qpack, kpack, vpack, attnout);

  // ---- output projection, fp32 epilogue straight to d_out
  gemm_bf16<<<dim3(NTOK/BM, N_EMBD/BN), 256, 0, stream>>>(attnout, wo, nullptr, out,
                                                          NTOK, N_EMBD, N_EMBD);
}
